// MultiHeadAttentionBlock_24111946400183
// MI455X (gfx1250) — matrix operand order repaired and checked by reading, not compiled, not run on hardware
//
#include <hip/hip_runtime.h>
#include <hip/hip_bf16.h>

#define B_ 4
#define S_ 2048
#define D_ 1024
#define H_ 16
#define DK_ 64
#define MTOT (B_ * S_)   // 8192

typedef __attribute__((ext_vector_type(16))) __bf16 bf16x16;
typedef __attribute__((ext_vector_type(8)))  float  f32x8;
typedef __attribute__((ext_vector_type(8)))  unsigned short ushort8;

union Frag16 { unsigned short u[16]; bf16x16 v; };

__device__ __forceinline__ unsigned short f2bf(float f) {
  unsigned int u = __float_as_uint(f);
  u += 0x7FFFu + ((u >> 16) & 1u);   // round-to-nearest-even
  return (unsigned short)(u >> 16);
}

__device__ __forceinline__ f32x8 zero8() {
  f32x8 z = {0.f, 0.f, 0.f, 0.f, 0.f, 0.f, 0.f, 0.f};
  return z;
}

__device__ __forceinline__ f32x8 wmma_bf16(bf16x16 a, bf16x16 b, f32x8 c) {
  // (neg_a, A, neg_b, B, c_mod, C, reuse_a, reuse_b)
  return __builtin_amdgcn_wmma_f32_16x16x32_bf16(false, a, false, b, (short)0, c, false, false);
}

// gfx1250 async copy: 16 bytes per lane, global -> LDS, tracked by ASYNCcnt.
// GVS addressing: mem = SADDR(sgpr64) + VADDR(vgpr32) ; VDST = LDS byte address.
__device__ __forceinline__ void async_copy_b128(unsigned lds_byte_addr,
                                                unsigned gl_byte_off,
                                                const void* base) {
  asm volatile("global_load_async_to_lds_b128 %0, %1, %2"
               :: "v"(lds_byte_addr), "v"(gl_byte_off),
                  "s"((unsigned long long)(uintptr_t)base)
               : "memory");
}

// --- Fragment loaders (bf16 sources only; fp32 is pre-converted once) -------
// A fragment (16x32): lane holds row M = lane%16.
// elements 0..7  -> K = k0 + (lane<16 ? 0 : 8) + i ; elements 8..15 -> +16
__device__ __forceinline__ bf16x16 load_a_bf16(const unsigned short* A, int ld, int m0, int k0) {
  int lane = threadIdx.x & 31;
  const unsigned short* p = A + (size_t)(m0 + (lane & 15)) * ld + k0 + ((lane >> 4) << 3);
  Frag16 f;
#pragma unroll
  for (int i = 0; i < 8; ++i) { f.u[i] = p[i]; f.u[8 + i] = p[16 + i]; }
  return f.v;
}
// B fragment (32x16): lane holds column N = n0 + lane%16; elems 0..7 -> K = k0 + (lane<16?0:8) + i ; elems 8..15 -> +16.
// Source "Bt" is row-major [n][k] (B column-major), leading dim ld over k.
__device__ __forceinline__ bf16x16 load_b_bf16(const unsigned short* Bt, int ld, int n0, int k0) {
  int lane = threadIdx.x & 31;
  const unsigned short* p = Bt + (size_t)(n0 + (lane & 15)) * ld + k0 + ((lane >> 4) << 3);
  Frag16 f;
#pragma unroll
  for (int i = 0; i < 8; ++i) { f.u[i] = p[i]; f.u[8 + i] = p[16 + i]; }
  return f.v;
}

// --- Kernel 0: streaming fp32 -> bf16 conversion (memory-bound, one pass) ---
__global__ __launch_bounds__(256)
void cvt_f32_bf16_kernel(const float* __restrict__ src, unsigned short* __restrict__ dst, int n) {
  int i = (blockIdx.x * 256 + threadIdx.x) * 8;
  if (i + 8 <= n) {
    float4 a = *(const float4*)(src + i);
    float4 b = *(const float4*)(src + i + 4);
    ushort8 o;
    o[0] = f2bf(a.x); o[1] = f2bf(a.y); o[2] = f2bf(a.z); o[3] = f2bf(a.w);
    o[4] = f2bf(b.x); o[5] = f2bf(b.y); o[6] = f2bf(b.z); o[7] = f2bf(b.w);
    *(ushort8*)(dst + i) = o;
  }
}

// --- Kernel 1: fused QKV projection: out = X @ W^T + b (all-bf16 operands) --
// blockIdx.z selects q/k/v. Q,K written bf16 [B,H,S,DK]; V written transposed [B,H,DK,S].
__global__ __launch_bounds__(128)
void qkv_proj_kernel(const unsigned short* Xq, const unsigned short* Xk, const unsigned short* Xv,
                     const unsigned short* Wq, const unsigned short* Wk, const unsigned short* Wv,
                     const float* bq, const float* bk, const float* bvv,
                     unsigned short* Qws, unsigned short* Kws, unsigned short* Vtws) {
  const int mode = blockIdx.z;
  const unsigned short* X; const unsigned short* W; const float* bias; unsigned short* Out;
  if (mode == 0)      { X = Xq; W = Wq; bias = bq;  Out = Qws;  }
  else if (mode == 1) { X = Xk; W = Wk; bias = bk;  Out = Kws;  }
  else                { X = Xv; W = Wv; bias = bvv; Out = Vtws; }

  const int lane = threadIdx.x & 31;
  const int wave = threadIdx.x >> 5;
  const int m0 = blockIdx.x * 64 + wave * 16;
  const int n0 = blockIdx.y * 64;

  f32x8 acc[4];
#pragma unroll
  for (int t = 0; t < 4; ++t) acc[t] = zero8();

  for (int k0 = 0; k0 < D_; k0 += 32) {
    if (k0 + 32 < D_)
      __builtin_prefetch(X + (size_t)m0 * D_ + k0 + 32, 0, 1);
    bf16x16 a = load_a_bf16(X, D_, m0, k0);
#pragma unroll
    for (int t = 0; t < 4; ++t) {
      bf16x16 b = load_b_bf16(W, D_, n0 + 16 * t, k0);
      acc[t] = wmma_bf16(a, b, acc[t]);
    }
  }

  const int rowbase = m0 + ((lane >> 4) << 3);
#pragma unroll
  for (int t = 0; t < 4; ++t) {
    int col = n0 + 16 * t + (lane & 15);
    int h = col >> 6, dk = col & 63;
    float bval = bias[col];
#pragma unroll
    for (int r = 0; r < 8; ++r) {
      int m = rowbase + r;
      int bb = m >> 11, s = m & (S_ - 1);
      float val = acc[t][r] + bval;
      size_t idx;
      if (mode < 2) idx = (((size_t)(bb * H_ + h)) * S_ + s) * DK_ + dk;
      else          idx = (((size_t)(bb * H_ + h)) * DK_ + dk) * S_ + s;
      Out[idx] = f2bf(val);
    }
  }
}

// --- Kernel 2: flash attention with double-buffered async K/V staging -------
// Dynamic LDS layout (bytes):
//   [    0,  8192)  pstage: 8 waves x (16x32 bf16) P-transpose tiles
//   [ 8192, 16384)  K chunk double buffer: 2 x [32 kv][64 dk] bf16
//   [16384, 24576)  V chunk double buffer: 2 x [64 dk][32 kv] bf16
__global__ __launch_bounds__(256)
void flash_attn_kernel(const unsigned short* Qws, const unsigned short* Kws,
                       const unsigned short* Vtws, const int* mask,
                       unsigned short* Ctx) {
  extern __shared__ unsigned short smem[];

  const int tid = threadIdx.x;
  const int lane = tid & 31;
  const int wave = tid >> 5;
  const int h = blockIdx.y, b = blockIdx.z;
  const int bh = b * H_ + h;
  const int m0 = blockIdx.x * 128 + wave * 16;     // query row within S

  const unsigned short* Qb = Qws + (size_t)bh * S_ * DK_;
  const unsigned short* Kb = Kws + (size_t)bh * S_ * DK_;
  const unsigned short* Vb = Vtws + (size_t)bh * DK_ * S_;
  unsigned short* ps = smem + wave * (16 * 32);

  bf16x16 qa0 = load_a_bf16(Qb, DK_, m0, 0);
  bf16x16 qa1 = load_a_bf16(Qb, DK_, m0, 32);

  f32x8 o[4];
#pragma unroll
  for (int t = 0; t < 4; ++t) o[t] = zero8();
  float rm[8], rl[8];
#pragma unroll
  for (int r = 0; r < 8; ++r) { rm[r] = -1e30f; rl[r] = 0.0f; }

  const float scale = 0.125f;  // 1/sqrt(DK)
  const int rowbase = m0 + ((lane >> 4) << 3);

  // Issue async DMA of one 32-wide K/V chunk into buffer `buf`.
  // K chunk: 4KB contiguous ([32][64] bf16). V chunk: 64 rows x 64B, row stride S_*2 bytes.
  auto issueChunk = [&](int kv0, int buf) {
    unsigned kdst = 8192u + (unsigned)buf * 4096u + (unsigned)tid * 16u;
    async_copy_b128(kdst, (unsigned)tid * 16u, Kb + (size_t)kv0 * DK_);
    unsigned vdst = 16384u + (unsigned)buf * 4096u +
                    (unsigned)((tid >> 2) * 64 + (tid & 3) * 16);
    unsigned voff = (unsigned)((tid >> 2) * (S_ * 2) + (tid & 3) * 16);
    async_copy_b128(vdst, voff, Vb + kv0);
  };

  issueChunk(0, 0);

  for (int kv0 = 0; kv0 < S_; kv0 += 32) {
    const int buf = (kv0 >> 5) & 1;
    __syncthreads();                          // all readers of buf^1 are done
    if (kv0 + 32 < S_) {
      issueChunk(kv0 + 32, buf ^ 1);          // overlap next DMA with compute
      asm volatile("s_wait_asynccnt 2" ::: "memory");   // current chunk (oldest 2) done
    } else {
      asm volatile("s_wait_asynccnt 0" ::: "memory");
    }
    __syncthreads();                          // chunk resident for every wave

    const unsigned short* Ksm = smem + 4096 + buf * 2048;  // [32 kv][64 dk], ld=64
    const unsigned short* Vsm = smem + 8192 + buf * 2048;  // [64 dk][32 kv], ld=32

    // scores: two 16x16 tiles, K-reduction over DK=64 in two WMMAs each
    f32x8 s0 = wmma_bf16(qa0, load_b_bf16(Ksm, DK_, 0, 0), zero8());
    s0 = wmma_bf16(qa1, load_b_bf16(Ksm, DK_, 0, 32), s0);
    f32x8 s1 = wmma_bf16(qa0, load_b_bf16(Ksm, DK_, 16, 0), zero8());
    s1 = wmma_bf16(qa1, load_b_bf16(Ksm, DK_, 16, 32), s1);

    const int col0 = kv0 + (lane & 15);
#pragma unroll
    for (int r = 0; r < 8; ++r) {
      int row = rowbase + r;
      float v0 = s0[r] * scale;
      float v1 = s1[r] * scale;
      if (mask[(size_t)row * S_ + col0] == 0)      v0 = -1e9f;
      if (mask[(size_t)row * S_ + col0 + 16] == 0) v1 = -1e9f;
      // row-wise max/sum across the 16 lanes of each half-wave
      float mx = fmaxf(v0, v1);
#pragma unroll
      for (int off = 1; off < 16; off <<= 1) mx = fmaxf(mx, __shfl_xor(mx, off, 32));
      float mnew = fmaxf(rm[r], mx);
      float alpha = __expf(rm[r] - mnew);
      float p0 = __expf(v0 - mnew);
      float p1 = __expf(v1 - mnew);
      float rs = p0 + p1;
#pragma unroll
      for (int off = 1; off < 16; off <<= 1) rs += __shfl_xor(rs, off, 32);
      rl[r] = rl[r] * alpha + rs;
      rm[r] = mnew;
#pragma unroll
      for (int t = 0; t < 4; ++t) o[t][r] *= alpha;
      // transpose P (C-layout -> A-layout) through wave-private LDS
      int prow = ((lane >> 4) << 3) + r;
      ps[prow * 32 + (lane & 15)]      = f2bf(p0);
      ps[prow * 32 + 16 + (lane & 15)] = f2bf(p1);
    }
    asm volatile("s_wait_dscnt 0" ::: "memory");   // LDS stores visible before A-frag loads
    bf16x16 pa = load_a_bf16(ps, 32, 0, 0);
#pragma unroll
    for (int t = 0; t < 4; ++t) {
      bf16x16 vb = load_b_bf16(Vsm, 32, t * 16, 0);   // V^T chunk: column dk contiguous over kv
      o[t] = wmma_bf16(pa, vb, o[t]);
    }
  }

  // normalize and write context in [B,S,D] (merged heads), bf16
#pragma unroll
  for (int r = 0; r < 8; ++r) {
    float inv = 1.0f / rl[r];
    int s = rowbase + r;
#pragma unroll
    for (int t = 0; t < 4; ++t) {
      int dcol = h * DK_ + t * 16 + (lane & 15);
      Ctx[((size_t)b * S_ + s) * D_ + dcol] = f2bf(o[t][r] * inv);
    }
  }
}

// --- Kernel 3: output projection: out = Ctx @ wo^T + bo (fp32 out) ----------
__global__ __launch_bounds__(128)
void out_proj_kernel(const unsigned short* Ctx, const unsigned short* Wo, const float* bo,
                     float* out) {
  const int lane = threadIdx.x & 31;
  const int wave = threadIdx.x >> 5;
  const int m0 = blockIdx.x * 64 + wave * 16;
  const int n0 = blockIdx.y * 64;

  f32x8 acc[4];
#pragma unroll
  for (int t = 0; t < 4; ++t) acc[t] = zero8();

  for (int k0 = 0; k0 < D_; k0 += 32) {
    bf16x16 a = load_a_bf16(Ctx, D_, m0, k0);
#pragma unroll
    for (int t = 0; t < 4; ++t) {
      bf16x16 bfrag = load_b_bf16(Wo, D_, n0 + 16 * t, k0);
      acc[t] = wmma_bf16(a, bfrag, acc[t]);
    }
  }

  const int rowbase = m0 + ((lane >> 4) << 3);
#pragma unroll
  for (int t = 0; t < 4; ++t) {
    int col = n0 + 16 * t + (lane & 15);
    float bval = bo[col];
#pragma unroll
    for (int r = 0; r < 8; ++r) {
      out[(size_t)(rowbase + r) * D_ + col] = acc[t][r] + bval;
    }
  }
}

// --- Host launch ------------------------------------------------------------
extern "C" void kernel_launch(void* const* d_in, const int* in_sizes, int n_in,
                              void* d_out, int out_size, void* d_ws, size_t ws_size,
                              hipStream_t stream) {
  const float* q  = (const float*)d_in[0];
  const float* k  = (const float*)d_in[1];
  const float* v  = (const float*)d_in[2];
  const int* mask = (const int*)d_in[3];
  const float* wq = (const float*)d_in[4];
  const float* bq = (const float*)d_in[5];
  const float* wk = (const float*)d_in[6];
  const float* bk = (const float*)d_in[7];
  const float* wv = (const float*)d_in[8];
  const float* bv = (const float*)d_in[9];
  const float* wo = (const float*)d_in[10];
  const float* bo = (const float*)d_in[11];

  const size_t nX = (size_t)MTOT * D_;   // 8,388,608 elements
  const size_t nW = (size_t)D_ * D_;     // 1,048,576 elements

  unsigned short* Xq   = (unsigned short*)d_ws;   // bf16 copies of fp32 operands
  unsigned short* Xk   = Xq + nX;
  unsigned short* Xv   = Xk + nX;
  unsigned short* Wqb  = Xv + nX;
  unsigned short* Wkb  = Wqb + nW;
  unsigned short* Wvb  = Wkb + nW;
  unsigned short* Wob  = Wvb + nW;
  unsigned short* Qws  = Wob + nW;                // bf16 [B,H,S,DK]
  unsigned short* Kws  = Qws + nX;                // bf16 [B,H,S,DK]
  unsigned short* Vtws = Kws + nX;                // bf16 [B,H,DK,S]
  unsigned short* Ctx  = Vtws + nX;               // bf16 [B,S,D]

  // one-time fp32 -> bf16 conversion (streaming, memory-bound)
  cvt_f32_bf16_kernel<<<(int)(nX / 8 / 256), 256, 0, stream>>>(q, Xq, (int)nX);
  cvt_f32_bf16_kernel<<<(int)(nX / 8 / 256), 256, 0, stream>>>(k, Xk, (int)nX);
  cvt_f32_bf16_kernel<<<(int)(nX / 8 / 256), 256, 0, stream>>>(v, Xv, (int)nX);
  cvt_f32_bf16_kernel<<<(int)(nW / 8 / 256), 256, 0, stream>>>(wq, Wqb, (int)nW);
  cvt_f32_bf16_kernel<<<(int)(nW / 8 / 256), 256, 0, stream>>>(wk, Wkb, (int)nW);
  cvt_f32_bf16_kernel<<<(int)(nW / 8 / 256), 256, 0, stream>>>(wv, Wvb, (int)nW);
  cvt_f32_bf16_kernel<<<(int)(nW / 8 / 256), 256, 0, stream>>>(wo, Wob, (int)nW);

  dim3 g1(MTOT / 64, D_ / 64, 3);
  qkv_proj_kernel<<<g1, 128, 0, stream>>>(Xq, Xk, Xv, Wqb, Wkb, Wvb, bq, bk, bv,
                                          Qws, Kws, Vtws);
  dim3 g2(S_ / 128, H_, B_);
  flash_attn_kernel<<<g2, 256, 24576, stream>>>(Qws, Kws, Vtws, mask, Ctx);
  dim3 g3(MTOT / 64, D_ / 64, 1);
  out_proj_kernel<<<g3, 128, 0, stream>>>(Ctx, Wob, bo, (float*)d_out);
}
